// CausalSelfAttention_69449621176580
// MI455X (gfx1250) — matrix operand order repaired and checked by reading, not compiled, not run on hardware
//
#include <hip/hip_runtime.h>
#include <hip/hip_fp16.h>

// ---------------------------------------------------------------------------
// CausalSelfAttention (GQA) for MI455X / gfx1250, wave32, WMMA f16->f32.
// B=2, T=2048, E=2048, H=16, D=128, KV heads G=4.
// Pipeline: convert/transpose -> Q/K/V GEMMs (TDM double-buffered) ->
//           V transpose -> flash attention -> output GEMM (f32 -> d_out).
// ---------------------------------------------------------------------------

typedef __attribute__((ext_vector_type(16))) _Float16 v16h;
typedef __attribute__((ext_vector_type(8)))  _Float16 v8h;
typedef __attribute__((ext_vector_type(8)))  float    v8f;

typedef unsigned int u32x4 __attribute__((ext_vector_type(4)));
typedef int          i32x8 __attribute__((ext_vector_type(8)));
typedef int          i32x4 __attribute__((ext_vector_type(4)));

typedef __attribute__((address_space(3))) const char lds_char;

__device__ __forceinline__ unsigned lds_off(const void* p) {
    // generic -> LDS addrspace cast; ptrtoint of as(3) ptr = byte offset in LDS
    return (unsigned)(unsigned long long)(lds_char*)p;
}

__device__ __forceinline__ v16h cat16(v8h lo, v8h hi) {
    return __builtin_shufflevector(lo, hi, 0, 1, 2, 3, 4, 5, 6, 7,
                                           8, 9, 10, 11, 12, 13, 14, 15);
}

__device__ __forceinline__ v8f wmma_f16(v16h a, v16h b, v8f c) {
    return __builtin_amdgcn_wmma_f32_16x16x32_f16(
        false, a, false, b, (short)0, c, false, false);
}

// ---- Tensor Data Mover: 2D tile (32 f16 cols x 128 rows) global -> LDS ----
// Pads +16B after each 64B row => LDS row stride = 40 halfwords.
__device__ __forceinline__ void tdm_load_tile(unsigned lds_byte,
                                              const _Float16* gptr,
                                              unsigned row_stride_elems) {
    unsigned long long ga = (unsigned long long)gptr;
    u32x4 g0;
    g0[0] = 1u;                                   // count=1, user mode
    g0[1] = lds_byte;                             // lds_addr
    g0[2] = (unsigned)(ga & 0xffffffffu);         // global_addr[31:0]
    g0[3] = (unsigned)((ga >> 32) & 0x01ffffffu)  // global_addr[56:32]
            | (2u << 30);                         // type = 2 ("image")
    i32x8 g1;
    // data_size=1 (2B), pad_enable, pad_interval=3 (16 DW), pad_amount=3 (4 DW)
    g1[0] = (int)((1u << 16) | (1u << 20) | (3u << 22) | (3u << 25));
    g1[1] = (int)(32u << 16);                     // tensor_dim0 = 32 (lo16)
    g1[2] = (int)(128u << 16);                    // tensor_dim1 = 128 (lo16)
    g1[3] = (int)(32u << 16);                     // tile_dim0 = 32
    g1[4] = (int)(128u);                          // tile_dim1 = 128, tile_dim2 = 0
    g1[5] = (int)row_stride_elems;                // tensor_dim0_stride lo32
    g1[6] = 0;
    g1[7] = 0;
    i32x4 z4 = {0, 0, 0, 0};
#if defined(__clang_major__) && __clang_major__ >= 23
    i32x8 z8 = {0, 0, 0, 0, 0, 0, 0, 0};
    __builtin_amdgcn_tensor_load_to_lds(g0, g1, z4, z4, z8, 0);
#else
    __builtin_amdgcn_tensor_load_to_lds(g0, g1, z4, z4, 0);
#endif
}

// ------------------------- elementwise conversions -------------------------

__global__ void cvt_f32_to_f16(const float* __restrict__ in,
                               _Float16* __restrict__ out, size_t n) {
    size_t i = (size_t)blockIdx.x * 256 + threadIdx.x;
    if (i < n) out[i] = (_Float16)in[i];
}

// in: f32 [K, N] row-major  ->  out: f16 [N, K] row-major (B^T for GEMM)
__global__ void tr_cvt_f32_to_f16(const float* __restrict__ in,
                                  _Float16* __restrict__ out,
                                  int K, int N, size_t total) {
    size_t i = (size_t)blockIdx.x * 256 + threadIdx.x;
    if (i >= total) return;
    int n = (int)(i % (size_t)N);
    int k = (int)(i / (size_t)N);
    out[(size_t)n * K + k] = (_Float16)in[i];
}

// V row-major [B*T, G*D] -> Vt [(b*G+g)*D + d][T]
__global__ void transpose_v_kernel(const _Float16* __restrict__ v,
                                   _Float16* __restrict__ vt, size_t total) {
    size_t i = (size_t)blockIdx.x * 256 + threadIdx.x;
    if (i >= total) return;
    int c = (int)(i % 512);
    size_t row = i / 512;
    int t = (int)(row % 2048);
    int b = (int)(row / 2048);
    vt[((size_t)b * 512 + c) * 2048 + t] = v[i];
}

// ------------------------------ WMMA GEMM ----------------------------------
// C[M,N] = A[M,K] * Bt[N,K]^T. Block tile 128x128, BK=32, double-buffered LDS
// fed by the Tensor Data Mover (wave 0 issues descriptors, TENSORcnt-gated).
// 8 waves (4 along M x 2 along N), each wave 2x4 frags of 16x16x32.

template <bool OUTF32>
__global__ __launch_bounds__(256)
void gemm_wmma(const _Float16* __restrict__ A,
               const _Float16* __restrict__ Bt,
               _Float16* __restrict__ Ch,
               float* __restrict__ Cf,
               int Mdim, int Ndim, int Kdim) {
    // 40-halfword row stride (TDM pads +4 DWORDs per 16 DWORDs stored)
    __shared__ __align__(16) _Float16 As[2][128 * 40];
    __shared__ __align__(16) _Float16 Bs[2][128 * 40];

    const int m0   = blockIdx.y * 128;
    const int n0   = blockIdx.x * 128;
    const int tid  = threadIdx.x;
    const int lane = tid & 31;
    const int wid  = tid >> 5;
    const int wm   = wid & 3;
    const int wn   = wid >> 2;
    const int lh   = lane & 15;
    const int sel  = lane >> 4;
    const bool issuer = (wid == 0);

    v8f acc[2][4] = {};

    const _Float16* Abase = A + (size_t)m0 * Kdim;
    const _Float16* Bbase = Bt + (size_t)n0 * Kdim;

    if (issuer) {
        tdm_load_tile(lds_off(&As[0][0]), Abase, (unsigned)Kdim);
        tdm_load_tile(lds_off(&Bs[0][0]), Bbase, (unsigned)Kdim);
    }

    const int nk = Kdim / 32;
    for (int ik = 0; ik < nk; ++ik) {
        const int buf = ik & 1;
        const bool has_next = (ik + 1) < nk;
        if (issuer) {
            if (has_next) {
                tdm_load_tile(lds_off(&As[buf ^ 1][0]),
                              Abase + (size_t)(ik + 1) * 32, (unsigned)Kdim);
                tdm_load_tile(lds_off(&Bs[buf ^ 1][0]),
                              Bbase + (size_t)(ik + 1) * 32, (unsigned)Kdim);
                __builtin_amdgcn_s_wait_tensorcnt(2);   // current tile done
            } else {
                __builtin_amdgcn_s_wait_tensorcnt(0);
            }
        }
        __syncthreads();   // tile `buf` visible to all waves

        v16h af[2];
        #pragma unroll
        for (int i = 0; i < 2; ++i) {
            int row = wm * 32 + i * 16 + lh;
            // A frag: lanes 0-15 K{0..7,16..23}; lanes 16-31 K{8..15,24..31}
            v8h lo = *(const v8h*)&As[buf][row * 40 + sel * 8];
            v8h hi = *(const v8h*)&As[buf][row * 40 + 16 + sel * 8];
            af[i] = cat16(lo, hi);
        }
        v16h bfr[4];
        #pragma unroll
        for (int j = 0; j < 4; ++j) {
            int row = wn * 64 + j * 16 + lh;
            // B frag: lanes 0-15 K{0..7,16..23}; lanes 16-31 K{8..15,24..31}
            v8h lo = *(const v8h*)&Bs[buf][row * 40 + sel * 8];
            v8h hi = *(const v8h*)&Bs[buf][row * 40 + 16 + sel * 8];
            bfr[j] = cat16(lo, hi);
        }

        #pragma unroll
        for (int i = 0; i < 2; ++i)
            #pragma unroll
            for (int j = 0; j < 4; ++j)
                acc[i][j] = wmma_f16(af[i], bfr[j], acc[i][j]);

        __syncthreads();   // all waves done with `buf` before TDM overwrites it
    }

    #pragma unroll
    for (int i = 0; i < 2; ++i)
        #pragma unroll
        for (int j = 0; j < 4; ++j)
            #pragma unroll
            for (int r = 0; r < 8; ++r) {
                int row = m0 + wm * 32 + i * 16 + sel * 8 + r;
                int col = n0 + wn * 64 + j * 16 + lh;
                float v = acc[i][j][r];
                if (OUTF32) Cf[(size_t)row * Ndim + col] = v;
                else        Ch[(size_t)row * Ndim + col] = (_Float16)v;
            }
    (void)Mdim;
}

// --------------------------- flash attention -------------------------------
// One wave per (b, h, 16-query tile). Online softmax over key blocks of 32.

__global__ __launch_bounds__(32)
void attn_flash(const _Float16* __restrict__ Q,
                const _Float16* __restrict__ Kc,
                const _Float16* __restrict__ Vt,
                _Float16* __restrict__ Y) {
    __shared__ __align__(16) _Float16 Ps[16 * 40];

    const int lane = threadIdx.x & 31;
    const int lh   = lane & 15;
    const int sel  = lane >> 4;

    const int blk = blockIdx.x;           // B*H*(T/16) = 4096
    const int q0  = (blk & 127) * 16;
    const int h   = (blk >> 7) & 15;
    const int b   = blk >> 11;
    const int g   = h >> 2;

    v16h qa[4];
    {
        const _Float16* qrow =
            Q + ((size_t)(b * 2048 + q0 + lh)) * 2048 + h * 128;
        #pragma unroll
        for (int kk = 0; kk < 4; ++kk) {
            v8h lo = *(const v8h*)&qrow[kk * 32 + sel * 8];
            v8h hi = *(const v8h*)&qrow[kk * 32 + 16 + sel * 8];
            qa[kk] = cat16(lo, hi);
        }
    }

    v8f o[8] = {};
    float mrow[8], lrow[8];
    #pragma unroll
    for (int r = 0; r < 8; ++r) { mrow[r] = -1.0e30f; lrow[r] = 0.0f; }

    const float scale = 0.08838834764831845f;   // 1/sqrt(128)
    const int nkb = ((q0 + 15) >> 5) + 1;

    for (int kb = 0; kb < nkb; ++kb) {
        const int t0 = kb * 32;

        v8f s[2] = {};
        #pragma unroll
        for (int sub = 0; sub < 2; ++sub) {
            #pragma unroll
            for (int kk = 0; kk < 4; ++kk) {
                const _Float16* krow =
                    Kc + ((size_t)(b * 2048 + t0 + sub * 16 + lh)) * 512 +
                    g * 128 + kk * 32 + sel * 8;
                v8h lo = *(const v8h*)&krow[0];
                v8h hi = *(const v8h*)&krow[16];
                s[sub] = wmma_f16(qa[kk], cat16(lo, hi), s[sub]);
            }
        }

        #pragma unroll
        for (int sub = 0; sub < 2; ++sub)
            #pragma unroll
            for (int r = 0; r < 8; ++r) {
                int key = t0 + sub * 16 + lh;
                int row = q0 + sel * 8 + r;
                float v = s[sub][r] * scale;
                s[sub][r] = (key <= row) ? v : -1.0e30f;
            }

        #pragma unroll
        for (int r = 0; r < 8; ++r) {
            float mx = fmaxf(s[0][r], s[1][r]);
            mx = fmaxf(mx, __shfl_xor(mx, 1, 32));
            mx = fmaxf(mx, __shfl_xor(mx, 2, 32));
            mx = fmaxf(mx, __shfl_xor(mx, 4, 32));
            mx = fmaxf(mx, __shfl_xor(mx, 8, 32));
            float mnew = fmaxf(mrow[r], mx);
            float sc = __expf(mrow[r] - mnew);
            float p0 = __expf(s[0][r] - mnew);
            float p1 = __expf(s[1][r] - mnew);
            Ps[(sel * 8 + r) * 40 + lh]      = (_Float16)p0;
            Ps[(sel * 8 + r) * 40 + 16 + lh] = (_Float16)p1;
            float sum = p0 + p1;
            sum += __shfl_xor(sum, 1, 32);
            sum += __shfl_xor(sum, 2, 32);
            sum += __shfl_xor(sum, 4, 32);
            sum += __shfl_xor(sum, 8, 32);
            lrow[r] = lrow[r] * sc + sum;
            mrow[r] = mnew;
            #pragma unroll
            for (int dt = 0; dt < 8; ++dt) o[dt][r] *= sc;
        }
        __syncthreads();

        v16h pf;
        {
            const _Float16* prow = &Ps[lh * 40];
            v8h lo = *(const v8h*)&prow[sel * 8];
            v8h hi = *(const v8h*)&prow[16 + sel * 8];
            pf = cat16(lo, hi);
        }

        #pragma unroll
        for (int dt = 0; dt < 8; ++dt) {
            const _Float16* vrow =
                Vt + (((size_t)(b * 4 + g)) * 128 + dt * 16 + lh) * 2048 +
                t0 + sel * 8;
            v8h lo = *(const v8h*)&vrow[0];
            v8h hi = *(const v8h*)&vrow[16];
            o[dt] = wmma_f16(pf, cat16(lo, hi), o[dt]);
        }
        __syncthreads();
    }

    #pragma unroll
    for (int dt = 0; dt < 8; ++dt)
        #pragma unroll
        for (int r = 0; r < 8; ++r) {
            int row = q0 + sel * 8 + r;
            float v = o[dt][r] / lrow[r];
            Y[((size_t)(b * 2048 + row)) * 2048 + h * 128 + dt * 16 + lh] =
                (_Float16)v;
        }
}

// ------------------------------- launcher ----------------------------------

extern "C" void kernel_launch(void* const* d_in, const int* in_sizes, int n_in,
                              void* d_out, int out_size, void* d_ws, size_t ws_size,
                              hipStream_t stream) {
    (void)in_sizes; (void)n_in; (void)out_size; (void)ws_size;
    const float* x  = (const float*)d_in[0];
    const float* Wq = (const float*)d_in[1];
    const float* Wk = (const float*)d_in[2];
    const float* Wv = (const float*)d_in[3];
    const float* Wo = (const float*)d_in[4];
    float* out = (float*)d_out;

    const int Bv = 2, T = 2048, E = 2048, G = 4, D = 128, H = 16;
    const size_t M  = (size_t)Bv * T;     // 4096
    const size_t KV = (size_t)G * D;      // 512

    char* ws = (char*)d_ws;
    size_t off = 0;
    auto alloc = [&](size_t elems) -> _Float16* {
        _Float16* p = (_Float16*)(ws + off);
        off += elems * sizeof(_Float16);
        return p;
    };
    _Float16* xh  = alloc(M * E);
    _Float16* WqT = alloc((size_t)E * E);
    _Float16* WkT = alloc(KV * E);
    _Float16* WvT = alloc(KV * E);
    _Float16* WoT = alloc((size_t)E * E);
    _Float16* Qh  = alloc(M * E);
    _Float16* Kh  = alloc(M * KV);
    _Float16* Vh  = alloc(M * KV);
    _Float16* Vt  = alloc(M * KV);
    _Float16* Yh  = alloc(M * E);

    {
        size_t n = M * E;
        cvt_f32_to_f16<<<dim3((unsigned)((n + 255) / 256)), dim3(256), 0, stream>>>(x, xh, n);
    }
    {
        size_t n = (size_t)E * E;
        tr_cvt_f32_to_f16<<<dim3((unsigned)((n + 255) / 256)), dim3(256), 0, stream>>>(Wq, WqT, E, E, n);
        tr_cvt_f32_to_f16<<<dim3((unsigned)((n + 255) / 256)), dim3(256), 0, stream>>>(Wo, WoT, E, E, n);
    }
    {
        size_t n = (size_t)E * KV;
        tr_cvt_f32_to_f16<<<dim3((unsigned)((n + 255) / 256)), dim3(256), 0, stream>>>(Wk, WkT, E, (int)KV, n);
        tr_cvt_f32_to_f16<<<dim3((unsigned)((n + 255) / 256)), dim3(256), 0, stream>>>(Wv, WvT, E, (int)KV, n);
    }

    gemm_wmma<false><<<dim3(E / 128, (unsigned)(M / 128)), dim3(256), 0, stream>>>(
        xh, WqT, Qh, nullptr, (int)M, E, E);
    gemm_wmma<false><<<dim3((unsigned)(KV / 128), (unsigned)(M / 128)), dim3(256), 0, stream>>>(
        xh, WkT, Kh, nullptr, (int)M, (int)KV, E);
    gemm_wmma<false><<<dim3((unsigned)(KV / 128), (unsigned)(M / 128)), dim3(256), 0, stream>>>(
        xh, WvT, Vh, nullptr, (int)M, (int)KV, E);

    {
        size_t n = M * KV;
        transpose_v_kernel<<<dim3((unsigned)((n + 255) / 256)), dim3(256), 0, stream>>>(Vh, Vt, n);
    }

    attn_flash<<<dim3(Bv * H * (T / 16)), dim3(32), 0, stream>>>(Qh, Kh, Vt, Yh);

    gemm_wmma<true><<<dim3(E / 128, (unsigned)(M / 128)), dim3(256), 0, stream>>>(
        Yh, WoT, nullptr, out, (int)M, E, E);
}
